// MoELayer_28750511079539
// MI455X (gfx1250) — matrix operand order repaired and checked by reading, not compiled, not run on hardware
//
#include <hip/hip_runtime.h>
#include <cstdint>
#include <cstddef>

// MoE top-2 FFN for MI455X (gfx1250): bf16 WMMA (16x16x32, f32 accum).
// Compute-bound (~412 GFLOP vs ~330MB traffic) -> bf16 matrix path + sparse
// top-2 routing. Weights/activations pre-converted to bf16 (weights
// pre-transposed to the WMMA B-fragment layout) once per launch when the
// workspace allows; steady-state staging is double-buffered
// GLOBAL_LOAD_ASYNC_TO_LDS_B128 with ASYNCcnt-threshold waits so global
// latency hides behind the current K-step's WMMAs.

#define D_DIM 1024
#define F_DIM 2048
#define NEXP  8
#define USE_ASYNC 1

typedef __bf16 v16bf __attribute__((ext_vector_type(16)));
typedef float  v8f   __attribute__((ext_vector_type(8)));

union Frag16 { v16bf v; uint4 q[2]; };

__device__ __forceinline__ unsigned short f32_bf16(float f) {
    unsigned int u = __float_as_uint(f);
    u += 0x7FFFu + ((u >> 16) & 1u);   // round-to-nearest-even
    return (unsigned short)(u >> 16);
}

// ---- async global->LDS 16B copy (CDNA5 §10.7 / §15.18 opcode 98) ----------
__device__ __forceinline__ void async_b128(void* lds_dst, const void* gsrc) {
#if USE_ASYNC
    unsigned ldsa = (unsigned)(uintptr_t)lds_dst;            // low 32b = LDS addr
    unsigned long long ga = (unsigned long long)(uintptr_t)gsrc;
    asm volatile("global_load_async_to_lds_b128 %0, %1, off"
                 :: "v"(ldsa), "v"(ga) : "memory");
#else
    *reinterpret_cast<uint4*>(lds_dst) = *reinterpret_cast<const uint4*>(gsrc);
#endif
}
// ASYNCcnt decrements in order for async loads -> waiting "<= N" with N
// next-step loads in flight guarantees the current buffer has landed.
template<int N>
__device__ __forceinline__ void async_wait() {
#if USE_ASYNC
    asm volatile("s_wait_asynccnt %0" :: "n"(N) : "memory");
#endif
}
__device__ __forceinline__ void ds_drain() {
    asm volatile("s_wait_dscnt 0x0" ::: "memory");
}

// ---------------------------------------------------------------- gating ----
__global__ void zero_meta(unsigned int* counts) {
    if (threadIdx.x < NEXP) counts[threadIdx.x] = 0u;
}

__global__ __launch_bounds__(256)
void moe_gate(const float* __restrict__ x, const float* __restrict__ gw,
              int* __restrict__ sel_e, float* __restrict__ sel_w,
              unsigned int* __restrict__ counts, int N) {
    int wave = threadIdx.x >> 5;
    int lane = threadIdx.x & 31;
    int t = blockIdx.x * 8 + wave;
    if (t >= N) return;
    const float* xr = x + (size_t)t * D_DIM;
    float acc[NEXP];
#pragma unroll
    for (int e = 0; e < NEXP; ++e) acc[e] = 0.f;
    for (int d = lane; d < D_DIM; d += 32) {
        float xv = xr[d];
#pragma unroll
        for (int e = 0; e < NEXP; ++e) acc[e] = fmaf(xv, gw[e * D_DIM + d], acc[e]);
    }
#pragma unroll
    for (int off = 16; off > 0; off >>= 1) {
#pragma unroll
        for (int e = 0; e < NEXP; ++e) acc[e] += __shfl_xor(acc[e], off, 32);
    }
    if (lane == 0) {
        float v1 = -3.4e38f, v2 = -3.4e38f; int i1 = 0, i2 = 0;
#pragma unroll
        for (int e = 0; e < NEXP; ++e) {
            float v = acc[e];
            if (v > v1) { v2 = v1; i2 = i1; v1 = v; i1 = e; }
            else if (v > v2) { v2 = v; i2 = e; }
        }
        float wa = 1.f / (1.f + __expf(v2 - v1));  // 2-way softmax == renorm top-k
        sel_e[2 * t]     = i1; sel_w[2 * t]     = wa;
        sel_e[2 * t + 1] = i2; sel_w[2 * t + 1] = 1.f - wa;
        atomicAdd(&counts[i1], 1u);
        atomicAdd(&counts[i2], 1u);
    }
}

__global__ void moe_scan(const unsigned int* __restrict__ counts,
                         unsigned int* __restrict__ offs,
                         unsigned int* __restrict__ fill) {
    offs[0] = 0u;
    for (int e = 0; e < NEXP; ++e) {
        offs[e + 1] = offs[e] + counts[e];
        fill[e] = offs[e];
    }
}

__global__ __launch_bounds__(256)
void moe_scatter(const int* __restrict__ sel_e, const float* __restrict__ sel_w,
                 unsigned int* __restrict__ fill,
                 int* __restrict__ perm_tok, float* __restrict__ perm_gate, int NS) {
    int i = blockIdx.x * 256 + threadIdx.x;
    if (i < NS) {
        int e = sel_e[i];
        unsigned int pos = atomicAdd(&fill[e], 1u);
        perm_tok[pos]  = i >> 1;
        perm_gate[pos] = sel_w[i];
    }
}

__global__ __launch_bounds__(256)
void zero_f32v4(float4* __restrict__ p, int n4) {
    int i = blockIdx.x * 256 + threadIdx.x;
    int stride = gridDim.x * 256;
    float4 z = {0.f, 0.f, 0.f, 0.f};
    for (; i < n4; i += stride) p[i] = z;
}

// ------------------------------------------------ bf16 pre-conversion pass --
__global__ __launch_bounds__(256)
void cvt_f32_bf16_v4(const float* __restrict__ src, unsigned short* __restrict__ dst,
                     int n4) {
    int i = blockIdx.x * 256 + threadIdx.x;
    int stride = gridDim.x * 256;
    for (; i < n4; i += stride) {
        float4 v = reinterpret_cast<const float4*>(src)[i];
        uint2 u;
        u.x = (unsigned)f32_bf16(v.x) | ((unsigned)f32_bf16(v.y) << 16);
        u.y = (unsigned)f32_bf16(v.z) | ((unsigned)f32_bf16(v.w) << 16);
        reinterpret_cast<uint2*>(dst)[i] = u;
    }
}

// src f32 [E][R][C] -> dst bf16 [E][C][R]; tiled 32x32 via LDS.
__global__ __launch_bounds__(256)
void transpose_cvt(const float* __restrict__ src, unsigned short* __restrict__ dst,
                   int R, int C) {
    __shared__ float tile[32][33];
    size_t eoff = (size_t)blockIdx.z * R * C;
    int c0 = blockIdx.x * 32, r0 = blockIdx.y * 32;
    int tx = threadIdx.x & 31, ty = threadIdx.x >> 5;
#pragma unroll
    for (int i = 0; i < 4; ++i) {
        int r = r0 + ty + i * 8;
        tile[ty + i * 8][tx] = src[eoff + (size_t)r * C + c0 + tx];
    }
    __syncthreads();
#pragma unroll
    for (int i = 0; i < 4; ++i) {
        int c = c0 + ty + i * 8;
        dst[eoff + (size_t)c * R + r0 + tx] = f32_bf16(tile[tx][ty + i * 8]);
    }
}

// ------------------------------------------------------------- GEMM pass 1 --
// H[r, f] = silu(X @ W1^T) * (X @ W3^T), bf16 out. 64x128 block tile,
// 8 waves of 2x2 WMMA tiles, K-step 32, double-buffered LDS.
// Tail rows are clamped to row ne-1 (results discarded at store), so every
// wave issues a constant number of async loads -> constant ASYNCcnt waits.
template<bool PREW>
__global__ __launch_bounds__(256)
void moe_gemm1(const float* __restrict__ x, const unsigned short* __restrict__ xb,
               const float* __restrict__ w1, const float* __restrict__ w3,
               const unsigned short* __restrict__ W1t, const unsigned short* __restrict__ W3t,
               const int* __restrict__ perm_tok, const unsigned int* __restrict__ offs,
               unsigned short* __restrict__ H, int e) {
    __shared__ unsigned short As [2][64 * 40];   // [m][k], row stride 40
    __shared__ unsigned short Bs1[2][32 * 136];  // [k][n], row stride 136 (lane%16=n)
    __shared__ unsigned short Bs3[2][32 * 136];
    __shared__ int tokS[64];

    unsigned int base = offs[e];
    unsigned int ne   = offs[e + 1] - base;
    unsigned int mb = blockIdx.x, nb = blockIdx.y;
    if (mb * 64u >= ne) return;

    int tid = threadIdx.x;
    if (tid < 64) {
        unsigned int r = mb * 64u + (unsigned int)tid;
        tokS[tid] = perm_tok[base + (r < ne ? r : ne - 1)];   // clamp tail
    }
    __syncthreads();

    int lane = tid & 31;
    int wm = (tid >> 5) & 1, wn = tid >> 6;
    int lrow = lane & 15, lhalf = lane >> 4;

    v8f zf = {0.f,0.f,0.f,0.f,0.f,0.f,0.f,0.f};
    v8f acc1[2][2], acc3[2][2];
#pragma unroll
    for (int i = 0; i < 2; ++i)
#pragma unroll
        for (int j = 0; j < 2; ++j) { acc1[i][j] = zf; acc3[i][j] = zf; }

    int ar = tid >> 2, ac = (tid & 3) * 8;        // A: one 16B chunk / thread
    int bk = tid >> 3, bn0 = (tid & 7) * 16;      // B (PREW): 2x16B / thread
    int brn = tid >> 1, brk = (tid & 1) * 16;     // B (fallback): 16 f32 / thread
    int tokA = tokS[ar];

    const float* W1f = w1 + (size_t)e * F_DIM * D_DIM;
    const float* W3f = w3 + (size_t)e * F_DIM * D_DIM;
    const unsigned short* W1b = W1t + (size_t)e * F_DIM * D_DIM;
    const unsigned short* W3b = W3t + (size_t)e * F_DIM * D_DIM;

    auto stage = [&](int buf, int k0) {
        if (PREW) {
            async_b128(&As[buf][ar * 40 + ac], xb + (size_t)tokA * D_DIM + k0 + ac);
            const unsigned short* p1 = W1b + (size_t)(k0 + bk) * F_DIM + nb * 128 + bn0;
            const unsigned short* p3 = W3b + (size_t)(k0 + bk) * F_DIM + nb * 128 + bn0;
            async_b128(&Bs1[buf][bk * 136 + bn0],     p1);
            async_b128(&Bs1[buf][bk * 136 + bn0 + 8], p1 + 8);
            async_b128(&Bs3[buf][bk * 136 + bn0],     p3);
            async_b128(&Bs3[buf][bk * 136 + bn0 + 8], p3 + 8);
        } else {
            const float4* p = reinterpret_cast<const float4*>(
                x + (size_t)tokA * D_DIM + k0 + ac);
            float4 u0 = p[0], u1 = p[1];
            float fa[8] = {u0.x,u0.y,u0.z,u0.w,u1.x,u1.y,u1.z,u1.w};
#pragma unroll
            for (int i = 0; i < 8; ++i) As[buf][ar * 40 + ac + i] = f32_bf16(fa[i]);
            const float* p1 = W1f + (size_t)(nb * 128 + brn) * D_DIM + k0 + brk;
            const float* p3 = W3f + (size_t)(nb * 128 + brn) * D_DIM + k0 + brk;
#pragma unroll
            for (int i = 0; i < 16; ++i) {
                Bs1[buf][(brk + i) * 136 + brn] = f32_bf16(p1[i]);
                Bs3[buf][(brk + i) * 136 + brn] = f32_bf16(p3[i]);
            }
        }
    };

    constexpr int ITERS = D_DIM / 32;
    stage(0, 0);
    for (int k = 0; k < ITERS; ++k) {
        int cur = k & 1;
        if (k + 1 < ITERS) {
            stage(cur ^ 1, (k + 1) * 32);            // prefetch next K-step
            if (PREW) async_wait<5>();               // current buffer landed
        } else {
            if (PREW) async_wait<0>();
        }
        __syncthreads();

        Frag16 af[2], b1[2], b3[2];
#pragma unroll
        for (int mi = 0; mi < 2; ++mi) {
            int off = (wm * 32 + mi * 16 + lrow) * 40 + lhalf * 8;
            af[mi].q[0] = *reinterpret_cast<const uint4*>(&As[cur][off]);
            af[mi].q[1] = *reinterpret_cast<const uint4*>(&As[cur][off + 16]);
        }
#pragma unroll
        for (int ni = 0; ni < 2; ++ni) {
            int off = wn * 32 + ni * 16 + lrow;
#pragma unroll
            for (int s = 0; s < 16; ++s) {
                int kk = (s < 8) ? (lhalf * 8 + s) : (16 + lhalf * 8 + (s - 8));
                reinterpret_cast<unsigned short*>(&b1[ni])[s] = Bs1[cur][kk * 136 + off];
                reinterpret_cast<unsigned short*>(&b3[ni])[s] = Bs3[cur][kk * 136 + off];
            }
        }
#pragma unroll
        for (int mi = 0; mi < 2; ++mi)
#pragma unroll
            for (int ni = 0; ni < 2; ++ni) {
                acc1[mi][ni] = __builtin_amdgcn_wmma_f32_16x16x32_bf16(
                    false, af[mi].v, false, b1[ni].v, (short)0, acc1[mi][ni], false, false);
                acc3[mi][ni] = __builtin_amdgcn_wmma_f32_16x16x32_bf16(
                    false, af[mi].v, false, b3[ni].v, (short)0, acc3[mi][ni], false, false);
            }
        ds_drain();          // our LDS reads done before next-iter async writes
        __syncthreads();
    }

#pragma unroll
    for (int mi = 0; mi < 2; ++mi)
#pragma unroll
        for (int ni = 0; ni < 2; ++ni)
#pragma unroll
            for (int j = 0; j < 8; ++j) {
                int m = wm * 32 + mi * 16 + lhalf * 8 + j;
                unsigned int r = mb * 64u + (unsigned int)m;
                if (r < ne) {
                    int fc = nb * 128 + wn * 32 + ni * 16 + lrow;
                    float a = acc1[mi][ni][j];
                    float hv = a * __builtin_amdgcn_rcpf(1.f + __expf(-a))
                               * acc3[mi][ni][j];
                    H[(size_t)r * F_DIM + fc] = f32_bf16(hv);
                }
            }
}

// ------------------------------------------------------------- GEMM pass 2 --
// out[tok, d] += gate * (H @ W2^T). Race-free: one slot per token per expert,
// experts serialized on the stream.
template<bool PREW>
__global__ __launch_bounds__(256)
void moe_gemm2(const unsigned short* __restrict__ H, const float* __restrict__ w2,
               const unsigned short* __restrict__ W2t,
               const int* __restrict__ perm_tok, const float* __restrict__ perm_gate,
               const unsigned int* __restrict__ offs, float* __restrict__ out, int e) {
    __shared__ unsigned short As[2][64 * 40];
    __shared__ unsigned short Bs[2][32 * 136];
    __shared__ int tokS[64];
    __shared__ float gateS[64];

    unsigned int base = offs[e];
    unsigned int ne   = offs[e + 1] - base;
    unsigned int mb = blockIdx.x, nb = blockIdx.y;
    if (mb * 64u >= ne) return;

    int tid = threadIdx.x;
    if (tid < 64) {
        unsigned int r = mb * 64u + (unsigned int)tid;
        tokS[tid]  = (r < ne) ? perm_tok[base + r]  : 0;
        gateS[tid] = (r < ne) ? perm_gate[base + r] : 0.f;
    }
    __syncthreads();

    int lane = tid & 31;
    int wm = (tid >> 5) & 1, wn = tid >> 6;
    int lrow = lane & 15, lhalf = lane >> 4;

    v8f zf = {0.f,0.f,0.f,0.f,0.f,0.f,0.f,0.f};
    v8f acc[2][2];
#pragma unroll
    for (int i = 0; i < 2; ++i)
#pragma unroll
        for (int j = 0; j < 2; ++j) acc[i][j] = zf;

    int ar = tid >> 2, ac = (tid & 3) * 8;
    int bk = tid >> 3, bn0 = (tid & 7) * 16;
    int brn = tid >> 1, brk = (tid & 1) * 16;
    unsigned int rA = mb * 64u + (unsigned int)ar;
    if (rA >= ne) rA = ne - 1;                     // clamp tail rows

    const float* W2f = w2 + (size_t)e * D_DIM * F_DIM;
    const unsigned short* W2b = W2t + (size_t)e * D_DIM * F_DIM;

    auto stage = [&](int buf, int k0) {
        async_b128(&As[buf][ar * 40 + ac],          // H already bf16 (both paths)
                   H + (size_t)rA * F_DIM + k0 + ac);
        if (PREW) {
            const unsigned short* p = W2b + (size_t)(k0 + bk) * D_DIM + nb * 128 + bn0;
            async_b128(&Bs[buf][bk * 136 + bn0],     p);
            async_b128(&Bs[buf][bk * 136 + bn0 + 8], p + 8);
        } else {
            const float* p = W2f + (size_t)(nb * 128 + brn) * F_DIM + k0 + brk;
#pragma unroll
            for (int i = 0; i < 16; ++i)
                Bs[buf][(brk + i) * 136 + brn] = f32_bf16(p[i]);
        }
    };

    constexpr int ITERS = F_DIM / 32;
    stage(0, 0);
    for (int k = 0; k < ITERS; ++k) {
        int cur = k & 1;
        if (k + 1 < ITERS) {
            stage(cur ^ 1, (k + 1) * 32);
            if (PREW) async_wait<3>(); else async_wait<1>();
        } else {
            async_wait<0>();
        }
        __syncthreads();

        Frag16 af[2], bf[2];
#pragma unroll
        for (int mi = 0; mi < 2; ++mi) {
            int off = (wm * 32 + mi * 16 + lrow) * 40 + lhalf * 8;
            af[mi].q[0] = *reinterpret_cast<const uint4*>(&As[cur][off]);
            af[mi].q[1] = *reinterpret_cast<const uint4*>(&As[cur][off + 16]);
        }
#pragma unroll
        for (int ni = 0; ni < 2; ++ni) {
            int off = wn * 32 + ni * 16 + lrow;
#pragma unroll
            for (int s = 0; s < 16; ++s) {
                int kk = (s < 8) ? (lhalf * 8 + s) : (16 + lhalf * 8 + (s - 8));
                reinterpret_cast<unsigned short*>(&bf[ni])[s] = Bs[cur][kk * 136 + off];
            }
        }
#pragma unroll
        for (int mi = 0; mi < 2; ++mi)
#pragma unroll
            for (int ni = 0; ni < 2; ++ni)
                acc[mi][ni] = __builtin_amdgcn_wmma_f32_16x16x32_bf16(
                    false, af[mi].v, false, bf[ni].v, (short)0, acc[mi][ni], false, false);
        ds_drain();
        __syncthreads();
    }

#pragma unroll
    for (int mi = 0; mi < 2; ++mi)
#pragma unroll
        for (int ni = 0; ni < 2; ++ni)
#pragma unroll
            for (int j = 0; j < 8; ++j) {
                int m = wm * 32 + mi * 16 + lhalf * 8 + j;
                unsigned int r = mb * 64u + (unsigned int)m;
                if (r < ne) {
                    int dc = nb * 128 + wn * 32 + ni * 16 + lrow;
                    size_t oi = (size_t)tokS[m] * D_DIM + dc;
                    out[oi] += gateS[m] * acc[mi][ni][j];
                }
            }
}

// ------------------------------------------------------------------ launch --
extern "C" void kernel_launch(void* const* d_in, const int* in_sizes, int n_in,
                              void* d_out, int out_size, void* d_ws, size_t ws_size,
                              hipStream_t stream) {
    (void)n_in; (void)out_size;
    const float* x  = (const float*)d_in[0];
    const float* gw = (const float*)d_in[1];
    const float* w1 = (const float*)d_in[2];
    const float* w2 = (const float*)d_in[3];
    const float* w3 = (const float*)d_in[4];
    float* out = (float*)d_out;

    int N  = in_sizes[0] / D_DIM;   // tokens (B*T = 16384)
    int NS = 2 * N;                 // routed slots (top-2)

    char* w = (char*)d_ws;
    unsigned int* counts = (unsigned int*)w;
    unsigned int* offs   = counts + 8;
    unsigned int* fill   = offs + 16;
    int*   sel_e     = (int*)  (w + 256);
    float* sel_w     = (float*)(w + 256 + (size_t)NS * 4);
    int*   perm_tok  = (int*)  (w + 256 + (size_t)NS * 8);
    float* perm_gate = (float*)(w + 256 + (size_t)NS * 12);
    size_t pos = (256 + (size_t)NS * 16 + 255) & ~(size_t)255;
    unsigned short* H   = (unsigned short*)(w + pos); pos += (size_t)N * F_DIM * 2;
    unsigned short* xb  = (unsigned short*)(w + pos); pos += (size_t)N * D_DIM * 2;
    unsigned short* W1t = (unsigned short*)(w + pos); pos += (size_t)NEXP * F_DIM * D_DIM * 2;
    unsigned short* W3t = (unsigned short*)(w + pos); pos += (size_t)NEXP * F_DIM * D_DIM * 2;
    unsigned short* W2t = (unsigned short*)(w + pos); pos += (size_t)NEXP * D_DIM * F_DIM * 2;
    bool prew = ws_size >= pos;     // ~197MB: pre-converted bf16 path

    zero_meta<<<1, 32, 0, stream>>>(counts);
    moe_gate<<<dim3((N + 7) / 8), 256, 0, stream>>>(x, gw, sel_e, sel_w, counts, N);
    moe_scan<<<1, 1, 0, stream>>>(counts, offs, fill);
    moe_scatter<<<dim3((NS + 255) / 256), 256, 0, stream>>>(sel_e, sel_w, fill,
                                                            perm_tok, perm_gate, NS);
    zero_f32v4<<<dim3(1024), 256, 0, stream>>>((float4*)out, N * D_DIM / 4);

    if (prew) {
        cvt_f32_bf16_v4<<<dim3(2048), 256, 0, stream>>>(x, xb, N * D_DIM / 4);
        transpose_cvt<<<dim3(D_DIM/32, F_DIM/32, NEXP), 256, 0, stream>>>(w1, W1t, F_DIM, D_DIM);
        transpose_cvt<<<dim3(D_DIM/32, F_DIM/32, NEXP), 256, 0, stream>>>(w3, W3t, F_DIM, D_DIM);
        transpose_cvt<<<dim3(F_DIM/32, D_DIM/32, NEXP), 256, 0, stream>>>(w2, W2t, D_DIM, F_DIM);
        for (int e = 0; e < NEXP; ++e) {
            moe_gemm1<true><<<dim3(N / 64, F_DIM / 128), 256, 0, stream>>>(
                x, xb, w1, w3, W1t, W3t, perm_tok, offs, H, e);
            moe_gemm2<true><<<dim3(N / 64, D_DIM / 128), 256, 0, stream>>>(
                H, w2, W2t, perm_tok, perm_gate, offs, out, e);
        }
    } else {
        for (int e = 0; e < NEXP; ++e) {
            moe_gemm1<false><<<dim3(N / 64, F_DIM / 128), 256, 0, stream>>>(
                x, xb, w1, w3, W1t, W3t, perm_tok, offs, H, e);
            moe_gemm2<false><<<dim3(N / 64, D_DIM / 128), 256, 0, stream>>>(
                H, w2, W2t, perm_tok, perm_gate, offs, out, e);
        }
    }
}
